// SVDLinear_13769665151156
// MI455X (gfx1250) — matrix operand order repaired and checked by reading, not compiled, not run on hardware
//
#include <hip/hip_runtime.h>
#include <hip/hip_bf16.h>

typedef _Float16 v16h __attribute__((ext_vector_type(16)));
typedef float    v8f  __attribute__((ext_vector_type(8)));
typedef unsigned v8u  __attribute__((ext_vector_type(8)));

#define NN    256
#define NANG  32640
#define BATCH 131072
#define LDA   264   // padded LDS stride (halves): 132 dwords/row, 132%64=4 -> conflict-free frag gathers
#define TILES 4     // 64-row tiles per block
#define GRIDX (BATCH / (64 * TILES))   // 512 blocks

// ---------------------------------------------------------------------------
// CDNA5 async global->LDS helpers (ASYNCcnt path), guarded so compile is safe
// ---------------------------------------------------------------------------
__device__ __forceinline__ void async_copy16(void* lds, const void* g) {
#if __has_builtin(__builtin_amdgcn_global_load_async_to_lds_b128)
    typedef int vi4 __attribute__((vector_size(16)));
    typedef __attribute__((address_space(1))) vi4* gp_t;   // pointer to global int4
    typedef __attribute__((address_space(3))) vi4* lp_t;   // pointer to LDS int4
    __builtin_amdgcn_global_load_async_to_lds_b128((gp_t)g, (lp_t)lds, 0, 0);
#else
    *(uint4*)lds = *(const uint4*)g;
#endif
}

__device__ __forceinline__ void wait_async0() {
#if __has_builtin(__builtin_amdgcn_s_wait_asynccnt)
    __builtin_amdgcn_s_wait_asynccnt(0);
#else
    asm volatile("s_wait_asynccnt 0x0" ::: "memory");
#endif
}

// ---------------------------------------------------------------------------
// Kernel A: cos/sin precompute (c,s) pairs for the Givens chain
// ---------------------------------------------------------------------------
__global__ void cs_kernel(const float* __restrict__ ang, float2* __restrict__ cs, int n) {
    int i = blockIdx.x * blockDim.x + threadIdx.x;
    if (i < n) {
        float s, c;
        sincosf(ang[i], &s, &c);
        cs[i] = make_float2(c, s);
    }
}

// ---------------------------------------------------------------------------
// Kernel B: build orthogonal matrices via sequential Givens chain.
// grid = 16 blocks x 32 threads. Block b: matrix (b>>3), columns (b&7)*32..+31.
// Each lane owns one column (independent evolution); column lives in LDS.
// cos/sin staged through LDS in chunks so the serial FMA chain never stalls
// on global latency.
// ---------------------------------------------------------------------------
__global__ __launch_bounds__(32) void givens_kernel(
        const float2* __restrict__ csU, const float2* __restrict__ csV,
        float* __restrict__ Urow, float* __restrict__ Utr,
        float* __restrict__ Vrow, float* __restrict__ Vtr) {
    __shared__ float  col[NN * 32];     // 32 KB: col[row*32 + lane]
    __shared__ float2 csl[2048];        // 16 KB staging for (c,s)

    const int lane = threadIdx.x;       // 0..31
    const int b    = blockIdx.x;        // 0..15
    const float2* cs   = (b & 8) ? csV  : csU;
    float*        Mrow = (b & 8) ? Vrow : Urow;
    float*        Mtr  = (b & 8) ? Vtr  : Utr;
    const int c = (b & 7) * 32 + lane;  // global column index

    for (int r = 0; r < NN; ++r) col[r * 32 + lane] = (r == c) ? 1.0f : 0.0f;

    int i = 0, j = 1;
    float vi = col[0 * 32 + lane];
    float vnext = 0.0f;

    for (int k0 = 0; k0 < NANG; k0 += 2048) {
        int cnt = NANG - k0; if (cnt > 2048) cnt = 2048;
        for (int t = lane; t < cnt; t += 32) csl[t] = cs[k0 + t];
        __syncthreads();
        for (int t = 0; t < cnt; ++t) {
            float2 a  = csl[t];                       // LDS broadcast, off-chain
            float  vj = col[j * 32 + lane];           // address independent of chain
            float  nvj = fmaf(a.y, vi, a.x * vj);
            vi = fmaf(a.x, vi, -(a.y * vj));          // the only chained op
            col[j * 32 + lane] = nvj;
            if (j == i + 1) vnext = nvj;
            ++j;
            if (j == NN) { col[i * 32 + lane] = vi; ++i; j = i + 1; vi = vnext; }
        }
        __syncthreads();
    }

    for (int r = 0; r < NN; ++r) {
        float v = col[r * 32 + lane];
        Mrow[r * NN + c] = v;   // coalesced
        Mtr[c * NN + r]  = v;
    }
}

// ---------------------------------------------------------------------------
// Kernel C: fuse  WfT[n][k] = W'[k][n] = sum_m U[k][m]*S[m]*V[n][m], as f16.
// ---------------------------------------------------------------------------
__global__ __launch_bounds__(256) void fuse_kernel(
        const float* __restrict__ Utr, const float* __restrict__ Vrow,
        const float* __restrict__ S, _Float16* __restrict__ WfT) {
    __shared__ float sv[NN];
    const int n = blockIdx.x, k = threadIdx.x;
    sv[k] = S[k] * Vrow[n * NN + k];
    __syncthreads();
    float acc = 0.0f;
    #pragma unroll 8
    for (int m = 0; m < NN; ++m) acc = fmaf(Utr[m * NN + k], sv[m], acc);
    WfT[n * NN + k] = (_Float16)acc;
}

// ---------------------------------------------------------------------------
// Kernel D: y = x @ W' + bias (131072x256x256), f16 WMMA, f32 accumulate.
// Persistent-ish blocks: grid=512, each handles 4 x-tiles of 64 rows.
// W'^T staged once per block; x f32 tile double-buffered via async->LDS.
// LDS: 132K (W'^T, stride 264) + 33K (x f16) + 2x64K (x f32 dbuf) = 300 KB.
// ---------------------------------------------------------------------------
__device__ __forceinline__ v8f wmma_f16(v16h a, v16h b, v8f c) {
    return __builtin_amdgcn_wmma_f32_16x16x32_f16(false, a, false, b, (short)0, c, false, false);
}

// A fragment (16x32 f16): lane L holds M=L&15; VGPR p: K = (L>>4)*16 + p*2 + lo
__device__ __forceinline__ v16h load_a(const _Float16* xs, int row, int kb) {
    const _Float16* p = xs + row * LDA + kb;   // kb = kt*32 + (lane>>4)*16, mult of 16
    uint2 q0 = *(const uint2*)(p);
    uint2 q1 = *(const uint2*)(p + 4);
    uint2 q2 = *(const uint2*)(p + 8);
    uint2 q3 = *(const uint2*)(p + 12);
    v8u w; w[0]=q0.x; w[1]=q0.y; w[2]=q1.x; w[3]=q1.y; w[4]=q2.x; w[5]=q2.y; w[6]=q3.x; w[7]=q3.y;
    return __builtin_bit_cast(v16h, w);
}

// B fragment (32x16 f16): lane L holds N=L&15; VGPR p: K = (L>>4)*16 + p*2 + lo -> 16 contiguous halves
__device__ __forceinline__ v16h load_b(const _Float16* wsb, int n, int kb2) {
    const _Float16* p = wsb + n * LDA + kb2;   // kb2 = kt*32 + (lane>>4)*16, mult of 16
    uint4 r0 = *(const uint4*)(p);
    uint4 r1 = *(const uint4*)(p + 8);
    v8u w; w[0]=r0.x; w[1]=r0.y; w[2]=r0.z; w[3]=r0.w; w[4]=r1.x; w[5]=r1.y; w[6]=r1.z; w[7]=r1.w;
    return __builtin_bit_cast(v16h, w);
}

extern __shared__ char smem_raw[];

#define SM_WSB  0
#define SM_XS   (NN * LDA * 2)            // 135168
#define SM_XF0  (SM_XS + 64 * LDA * 2)    // 168960
#define SM_XF1  (SM_XF0 + 65536)          // 234496
#define SM_TOT  (SM_XF1 + 65536)          // 300032

__global__ __launch_bounds__(256, 1) void gemm_kernel(
        const float* __restrict__ x, const _Float16* __restrict__ Wf,
        const float* __restrict__ bias, float* __restrict__ y) {
    _Float16* wsb = (_Float16*)(smem_raw + SM_WSB);
    _Float16* xs  = (_Float16*)(smem_raw + SM_XS);
    float*    xf0 = (float*)(smem_raw + SM_XF0);
    float*    xf1 = (float*)(smem_raw + SM_XF1);
    const int tid = threadIdx.x;
    const int rowBase0 = blockIdx.x * (64 * TILES);

    // async stage W'^T (f16) global -> LDS, restriding 256 -> LDA (per-lane LDS dest)
    #pragma unroll
    for (int it = 0; it < 32; ++it) {
        int idx = it * 256 + tid;                       // 0..8191 (16B chunks)
        int n = idx >> 5, c8 = idx & 31;
        async_copy16(wsb + n * LDA + c8 * 8, (const char*)Wf + idx * 16);
    }
    // async load x tile 0 (raw f32)
    #pragma unroll
    for (int it = 0; it < 16; ++it) {
        int idx = it * 256 + tid;
        async_copy16(xf0 + idx * 4, x + rowBase0 * NN + idx * 4);
    }
    wait_async0();
    __syncthreads();

    const int lane  = tid & 31;
    const int wave  = tid >> 5;
    const int waveM = (wave >> 2) * 32;      // 0 or 32
    const int waveN = (wave & 3) * 64;       // 0,64,128,192
    const int ln = lane & 15, g = lane >> 4;

    for (int t = 0; t < TILES; ++t) {
        const float* cur = (t & 1) ? xf1 : xf0;
        float*       nxt = (t & 1) ? xf0 : xf1;
        const int rowBase = rowBase0 + t * 64;

        // prefetch next f32 tile into the other buffer (overlaps convert+compute)
        if (t + 1 < TILES) {
            #pragma unroll
            for (int it = 0; it < 16; ++it) {
                int idx = it * 256 + tid;
                async_copy16(nxt + idx * 4, x + (rowBase + 64) * NN + idx * 4);
            }
        }

        // convert cur f32 (LDS) -> xs f16 (LDS, padded stride)
        #pragma unroll
        for (int it = 0; it < 16; ++it) {
            int idx = it * 256 + tid;                  // 0..4095 (float4 chunks)
            int r = idx >> 6, c4 = idx & 63;
            float4 f = *(const float4*)(cur + idx * 4);
            _Float16 h0 = (_Float16)f.x, h1 = (_Float16)f.y, h2 = (_Float16)f.z, h3 = (_Float16)f.w;
            uint2 pk;
            pk.x = ((unsigned)__builtin_bit_cast(unsigned short, h1) << 16) | __builtin_bit_cast(unsigned short, h0);
            pk.y = ((unsigned)__builtin_bit_cast(unsigned short, h3) << 16) | __builtin_bit_cast(unsigned short, h2);
            *(uint2*)(xs + r * LDA + c4 * 4) = pk;
        }
        __syncthreads();

        v8f acc[2][4];
        #pragma unroll
        for (int m = 0; m < 2; ++m)
            #pragma unroll
            for (int nt = 0; nt < 4; ++nt) acc[m][nt] = v8f{0,0,0,0,0,0,0,0};

        for (int kt = 0; kt < 8; ++kt) {
            const int kb  = kt * 32 + (g << 4);
            const int kb2 = kt * 32 + (g << 4);
            v16h a0 = load_a(xs, waveM + ln,      kb);
            v16h a1 = load_a(xs, waveM + 16 + ln, kb);
            #pragma unroll
            for (int nt = 0; nt < 4; ++nt) {
                v16h bf = load_b(wsb, waveN + nt * 16 + ln, kb2);
                acc[0][nt] = wmma_f16(a0, bf, acc[0][nt]);
                acc[1][nt] = wmma_f16(a1, bf, acc[1][nt]);
            }
        }

        // epilogue: C/D layout -> y (f32, non-temporal) + bias
        #pragma unroll
        for (int m = 0; m < 2; ++m) {
            const int rb = rowBase + waveM + m * 16 + g * 8;
            #pragma unroll
            for (int nt = 0; nt < 4; ++nt) {
                const int ncol = waveN + nt * 16 + ln;
                const float bv = bias[ncol];
                #pragma unroll
                for (int r = 0; r < 8; ++r)
                    __builtin_nontemporal_store(acc[m][nt][r] + bv, &y[(rb + r) * NN + ncol]);
            }
        }

        if (t + 1 < TILES) wait_async0();
        __syncthreads();   // xs / cur reusable next iteration
    }
}

// ---------------------------------------------------------------------------
// Host launcher
// ---------------------------------------------------------------------------
extern "C" void kernel_launch(void* const* d_in, const int* in_sizes, int n_in,
                              void* d_out, int out_size, void* d_ws, size_t ws_size,
                              hipStream_t stream) {
    const float* x    = (const float*)d_in[0];
    const float* Uw   = (const float*)d_in[1];
    const float* Vw   = (const float*)d_in[2];
    const float* Sw   = (const float*)d_in[3];
    const float* bias = (const float*)d_in[4];
    float* y = (float*)d_out;

    char* ws = (char*)d_ws;
    float2*   csU  = (float2*)(ws + 0);
    float2*   csV  = (float2*)(ws + 262144);
    float*    Urow = (float*)(ws + 524288);
    float*    Utr  = (float*)(ws + 786432);
    float*    Vrow = (float*)(ws + 1048576);
    float*    Vtr  = (float*)(ws + 1310720);
    _Float16* WfT  = (_Float16*)(ws + 1572864);

    cs_kernel<<<(NANG + 255) / 256, 256, 0, stream>>>(Uw, csU, NANG);
    cs_kernel<<<(NANG + 255) / 256, 256, 0, stream>>>(Vw, csV, NANG);

    givens_kernel<<<16, 32, 0, stream>>>(csU, csV, Urow, Utr, Vrow, Vtr);

    fuse_kernel<<<NN, 256, 0, stream>>>(Utr, Vrow, Sw, WfT);

    (void)hipFuncSetAttribute((const void*)gemm_kernel,
                              hipFuncAttributeMaxDynamicSharedMemorySize, SM_TOT);
    gemm_kernel<<<GRIDX, 256, SM_TOT, stream>>>(x, WfT, bias, y);
}
